// Generator_KP_48137993454126
// MI455X (gfx1250) — matrix operand order repaired and checked by reading, not compiled, not run on hardware
//
#include <hip/hip_runtime.h>
#include <hip/hip_bf16.h>
#include <math.h>

// ---------------------------------------------------------------------------
// Problem constants
// ---------------------------------------------------------------------------
#define KD   121      // flow dimension (11*11)
#define HID  15
#define NB   5
#define EPSB 1e-5f
#define ALPH 1e-6f
#define IW   4096     // input H=W
#define OW   2043     // output H=W : (4096-11)/2 + 1

typedef _Float16 v16h __attribute__((ext_vector_type(16)));
typedef float    v8f  __attribute__((ext_vector_type(8)));

// ---------------------------------------------------------------------------
// Kernel 1: MAF inverse (5 MADE blocks, 605 strictly-sequential autoregressive
// steps) + post-process -> 11x11 kernel. Latency-bound: one wave, incremental
// hidden-state update makes each step O(HID) instead of O(HID*D).
// sum_ldj / logprob are dead w.r.t. the outputs and are skipped.
// ---------------------------------------------------------------------------
__global__ __launch_bounds__(32)
void flow_inverse_kernel(const float* __restrict__ kc,
                         const float* __restrict__ W1, const float* __restrict__ b1,
                         const float* __restrict__ W2, const float* __restrict__ b2,
                         const float* __restrict__ lg, const float* __restrict__ bt,
                         const float* __restrict__ mn, const float* __restrict__ vr,
                         float* __restrict__ out_k, float* __restrict__ ws_k)
{
    __shared__ float sx[KD];            // current x
    __shared__ float su[KD];            // u for current MADE block
    __shared__ float sW1[HID * KD];     // W1[b]  (HID x D)
    __shared__ float sW2[2 * KD * HID]; // W2[b]  (2D x HID)

    const int lane = threadIdx.x;

    for (int d = lane; d < KD; d += 32) sx[d] = kc[d];
    __syncthreads();

    for (int b = NB - 1; b >= 0; --b) {
        const float* W1b = W1 + b * HID * KD;
        const float* W2b = W2 + b * 2 * KD * HID;
        for (int i = lane; i < HID * KD; i += 32)     sW1[i] = W1b[i];
        for (int i = lane; i < 2 * KD * HID; i += 32) sW2[i] = W2b[i];

        // BatchNorm inverse (eval): x = (x-beta)*exp(-lg)*sqrt(var+eps)+mean
        for (int d = lane; d < KD; d += 32) {
            float v = sx[d];
            v = (v - bt[b * KD + d]) * expf(-lg[b * KD + d]) *
                sqrtf(vr[b * KD + d] + EPSB) + mn[b * KD + d];
            su[d] = v;
        }
        __syncthreads();

        // MADE inverse; processing follows deg so step t has degree t.
        float acc = (lane < HID) ? b1[b * HID + lane] : 0.0f;
        for (int t = 0; t < KD; ++t) {
            const int i = ((b & 1) == 0) ? t : (KD - 1 - t);
            const float h = fmaxf(acc, 0.0f);
            float p = 0.0f, q = 0.0f;
            if (lane < HID && t > lane) {           // m2 strict: deg_in[i] > j
                p = h * sW2[i * HID + lane];        // row i    -> m
                q = h * sW2[(KD + i) * HID + lane]; // row D+i  -> loga
            }
            #pragma unroll
            for (int o = 16; o > 0; o >>= 1) {      // wave32 butterfly
                p += __shfl_xor(p, o, 32);
                q += __shfl_xor(q, o, 32);
            }
            const float m_i  = p + b2[b * 2 * KD + i];
            const float loga = q + b2[b * 2 * KD + KD + i];
            const float v = su[i] * expf(loga) + m_i;
            if (lane == 0) sx[i] = v;
            if (lane < HID && lane >= t)            // m1: dh[j] >= deg_in[i]
                acc += sW1[lane * KD + i] * v;
        }
        __syncthreads();
    }

    // post-process: inverse-logit, normalize to sum 1
    float vals[4];
    float s = 0.0f;
    int c = 0;
    for (int d = lane; d < KD; d += 32, ++c) {
        const float sg = 1.0f / (1.0f + expf(-sx[d]));
        const float v  = (sg - ALPH) / (1.0f - 2.0f * ALPH);
        vals[c] = v;
        s += v;
    }
    #pragma unroll
    for (int o = 16; o > 0; o >>= 1) s += __shfl_xor(s, o, 32);
    const float inv = 1.0f / s;
    c = 0;
    for (int d = lane; d < KD; d += 32, ++c) {
        const float v = vals[c] * inv;
        out_k[d] = v;
        ws_k[d]  = v;
    }
}

// ---------------------------------------------------------------------------
// Kernel 2: depthwise 11x11 stride-2 conv as implicit GEMM on the WMMA pipe.
//
// Per input row r:  A[m,k] = in[r][2(x0+m)+k]   (independent of n)
//                   B[k,n] = w[r-2(y0+n)][k]    (ky encoded per output row)
// Accumulating C[m,n] over the 41-row band gives the exact 16x16 output tile.
// Two input rows packed per v_wmma_f32_16x16x32_f16 (K 0..15 -> row r,
// K 16..31 -> row r+1) => 21 WMMAs/tile. f16 A/B, f32 accumulate.
//
// Staging: interior blocks stream the f32 tile into LDS with
// global_load_async_to_lds_b128 (ASYNCcnt); edge blocks take a guarded path.
// Block = 256 threads = 8 waves = 32(y) x 64(x) outputs.
// ---------------------------------------------------------------------------
#define TILE_X   64
#define TILE_Y   32
#define LDS_COLS 144   // 2*64 + 16 (covers padded A reads); 36 float4/row
#define LDS_ROWS 74    // 2*32 + 10, +1 pad row for the r+1 read of the pair

__global__ __launch_bounds__(256)
void conv_wmma_kernel(const float* __restrict__ in,
                      const float* __restrict__ kf,   // 121 f32 kernel taps
                      float* __restrict__ out)
{
    __shared__ float    stile[LDS_ROWS][LDS_COLS];    // 42.6 KB, f32
    __shared__ _Float16 swf[11][16];                  // kx padded to 16 with 0

    const int tid = threadIdx.x;
    const int ch  = blockIdx.z;
    const int X0  = blockIdx.x * TILE_X;
    const int Y0  = blockIdx.y * TILE_Y;
    const float* inc = in + (size_t)ch * IW * IW;

    if (tid < 176) {
        const int ky = tid >> 4, kx = tid & 15;
        const float w = (kx < 11) ? kf[ky * 11 + kx] : 0.0f;
        swf[ky][kx] = (_Float16)w;
    }

    const int r0 = 2 * Y0, c0 = 2 * X0;
    const bool interior = (r0 + LDS_ROWS <= IW) && (c0 + LDS_COLS <= IW);
    if (interior) {
        // Async DMA: 16B per lane per op, global -> LDS, no VGPR round-trip.
        // Low 32 bits of a generic shared pointer are the LDS byte offset
        // (ISA: flat->LDS uses addr[31:0]).
        const unsigned lds0 = (unsigned)(size_t)&stile[0][0];
        const float* gbase  = inc + (size_t)r0 * IW + c0;   // 16B aligned (c0 % 128 == 0)
        for (int cidx = tid; cidx < LDS_ROWS * (LDS_COLS / 4); cidx += 256) {
            const int lr  = cidx / (LDS_COLS / 4);
            const int lc4 = cidx % (LDS_COLS / 4);
            const unsigned laddr = lds0 + (unsigned)(lr * LDS_COLS + lc4 * 4) * 4u;
            const unsigned long long gaddr =
                (unsigned long long)(size_t)(gbase + (size_t)lr * IW + lc4 * 4);
            asm volatile("global_load_async_to_lds_b128 %0, %1, off"
                         :: "v"(laddr), "v"(gaddr) : "memory");
        }
        asm volatile("s_wait_asynccnt 0x0" ::: "memory");
    } else {
        for (int idx = tid; idx < LDS_ROWS * LDS_COLS; idx += 256) {
            const int lr = idx / LDS_COLS, lc = idx % LDS_COLS;
            const int gr = r0 + lr, gc = c0 + lc;
            stile[lr][lc] = (gr < IW && gc < IW) ? inc[(size_t)gr * IW + gc] : 0.0f;
        }
    }
    __syncthreads();

    const int wave = tid >> 5;
    const int lane = tid & 31;
    const int wy   = wave >> 2;            // 0..1
    const int wx   = wave & 3;             // 0..3
    const int y0   = Y0 + wy * 16;
    const int x0   = X0 + wx * 16;
    const int lrb  = 32 * wy;              // LDS row base (r_off = 0)
    const int lcb  = 32 * wx;              // LDS col base
    const int mloc = lane & 15;
    const bool hi  = lane >= 16;

    v8f c = {};
    #pragma unroll 3
    for (int r_off = 0; r_off <= 40; r_off += 2) {
        const int lr = lrb + r_off;
        // A: lane<16: e0..15 -> K0..15 (row r taps 0..10, rest 0);
        // lane>=16: e0..15 -> K16..31 (row r+1 taps 0..10, rest 0).
        v16h a;
        const int cb = lcb + 2 * mloc;
        const int ar = lr + (hi ? 1 : 0);
        #pragma unroll
        for (int e = 0; e < 16; ++e)
            a[e] = (e < 11) ? (_Float16)stile[ar][cb + e] : (_Float16)0.0f;
        // B: lane<16 carries K0..15 (row r) for column n; lane>=16 carries
        // K16..31 (row r+1). Zero unless ky = r - 2(y0+n) lands in [0,10].
        v16h bm = {};
        const int ky = (hi ? r_off + 1 : r_off) - 2 * mloc;
        if (ky >= 0 && ky <= 10) {
            #pragma unroll
            for (int e = 0; e < 11; ++e) bm[e] = swf[ky][e];
        }
        c = __builtin_amdgcn_wmma_f32_16x16x32_f16(
                false, a, false, bm, (short)0, c, false, false);
    }

    // C/D layout: value(vgpr v, lane l) = C[M = v + 8*(l>=16)][N = l&15]
    const int yb = y0 + mloc;
    if (yb < OW) {
        float* op = out + (size_t)ch * OW * OW + (size_t)yb * OW;
        #pragma unroll
        for (int v = 0; v < 8; ++v) {
            const int x = x0 + v + (hi ? 8 : 0);
            if (x < OW) op[x] = c[v];
        }
    }
}

// ---------------------------------------------------------------------------
// Launch: kernel1 (1 wave, sequential flow inverse) -> kernel2 (32x64x3
// blocks), ordered on `stream`. d_out = [out_put: 3*2043*2043 f32][out_k: 121
// f32]. d_ws carries the 121 post-processed taps from kernel1 to kernel2.
// ---------------------------------------------------------------------------
extern "C" void kernel_launch(void* const* d_in, const int* in_sizes, int n_in,
                              void* d_out, int out_size, void* d_ws, size_t ws_size,
                              hipStream_t stream)
{
    const float* in_t = (const float*)d_in[0];
    const float* kc   = (const float*)d_in[1];
    const float* W1   = (const float*)d_in[2];
    const float* b1   = (const float*)d_in[3];
    const float* W2   = (const float*)d_in[4];
    const float* b2   = (const float*)d_in[5];
    const float* lg   = (const float*)d_in[6];
    const float* bt   = (const float*)d_in[7];
    const float* mn   = (const float*)d_in[8];
    const float* vr   = (const float*)d_in[9];

    float* out   = (float*)d_out;
    float* out_k = out + (size_t)3 * OW * OW;
    float* ws_k  = (float*)d_ws;

    flow_inverse_kernel<<<1, 32, 0, stream>>>(kc, W1, b1, W2, b2,
                                              lg, bt, mn, vr, out_k, ws_k);

    dim3 grid((OW + TILE_X - 1) / TILE_X,   // 32
              (OW + TILE_Y - 1) / TILE_Y,   // 64
              3);
    conv_wmma_kernel<<<grid, 256, 0, stream>>>(in_t, ws_k, out);
}
